// GroupedQueryAttention_46196668236315
// MI455X (gfx1250) — matrix operand order repaired and checked by reading, not compiled, not run on hardware
//
#include <hip/hip_runtime.h>
#include <cstdint>
#include <cstddef>

// Problem constants (match reference)
#define BB   2
#define SS   2048
#define DD   2048
#define HH   32
#define KVH  8
#define HDIM 64
#define MROWS (BB * SS)   // 4096

typedef __bf16 bf16_t;
typedef bf16_t bfx16 __attribute__((ext_vector_type(16)));
typedef float  v8f   __attribute__((ext_vector_type(8)));

union FragBF {
    bfx16 v;
    unsigned short u[16];
    uint4 q[2];
};

__device__ inline v8f zero8() {
    v8f v;
#pragma unroll
    for (int i = 0; i < 8; ++i) v[i] = 0.0f;
    return v;
}

// fp32 -> bf16 (round-to-nearest-even)
__device__ inline unsigned short f2bf(float f) {
    unsigned int u = __float_as_uint(f);
    u += 0x7FFFu + ((u >> 16) & 1u);
    return (unsigned short)(u >> 16);
}

__device__ inline v8f wmma_bf16(bfx16 a, bfx16 b, v8f c) {
    return __builtin_amdgcn_wmma_f32_16x16x32_bf16(false, a, false, b, (short)0, c,
                                                   false, false);
}

// A-operand fragment (16x32 bf16) from row-major source; rows = tile rows.
// lanes 0-15 row=lane, K={0..7}+{16..23}; lanes 16-31 row=lane-16, K={8..15}+{24..31}.
__device__ inline bfx16 load_frag_a(const unsigned short* base, int ld, int k0) {
    const int lane = threadIdx.x & 31;
    const int r  = lane & 15;
    const int kb = k0 + ((lane >> 4) << 3);
    const unsigned short* p = base + (size_t)r * ld + kb;
    FragBF f;
    f.q[0] = *(const uint4*)(p);
    f.q[1] = *(const uint4*)(p + 16);
    return f.v;
}

// B-operand fragment (32x16) where the WMMA K-dim is CONTIGUOUS per lane
// (source laid out [n][k]).  lanes 0-15: col=lane, K={0..7}+{16..23}; lanes 16-31: K={8..15}+{24..31}.
__device__ inline bfx16 load_frag_b_kcontig(const unsigned short* base, int ld,
                                            int n0, int k0) {
    const int lane = threadIdx.x & 31;
    const int n  = n0 + (lane & 15);
    const int kb = k0 + ((lane >> 4) << 3);
    const unsigned short* p = base + (size_t)n * ld + kb;
    FragBF f;
    f.q[0] = *(const uint4*)(p);
    f.q[1] = *(const uint4*)(p + 16);
    return f.v;
}

// ---------------- elementwise kernels ----------------

__global__ void conv_f32_bf16_kernel(const float* __restrict__ X,
                                     unsigned short* __restrict__ Y, int n) {
    int i = blockIdx.x * blockDim.x + threadIdx.x;
    if (i < n) Y[i] = f2bf(X[i]);
}

// RoPE on interleaved (even,odd) pairs + convert to bf16.
__global__ void rope_bf16_kernel(const float* __restrict__ X,
                                 const float* __restrict__ Ct,
                                 const float* __restrict__ St,
                                 unsigned short* __restrict__ Y,
                                 int nh, int total_pairs) {
    int idx = blockIdx.x * blockDim.x + threadIdx.x;
    if (idx >= total_pairs) return;
    const int hd2 = HDIM / 2;
    const int p = idx % hd2;
    const int rest = idx / hd2;              // (b*S+s)*nh + h
    const int s = (rest / nh) % SS;
    const float c  = Ct[s * hd2 + p];
    const float si = St[s * hd2 + p];
    const float x1 = X[(size_t)idx * 2 + 0];
    const float x2 = X[(size_t)idx * 2 + 1];
    Y[(size_t)idx * 2 + 0] = f2bf(x1 * c - x2 * si);
    Y[(size_t)idx * 2 + 1] = f2bf(x1 * si + x2 * c);
}

// Fused f32->bf16 + transpose of V: Vf[B,S,KV,HD] -> Vt[B,KV,HD,S]
__global__ void __launch_bounds__(256)
transpose_v_kernel(const float* __restrict__ Vf, unsigned short* __restrict__ Vt) {
    __shared__ float tile[32][33];
    const int s0 = blockIdx.x * 32;
    const int d0 = blockIdx.y * 32;
    const int bk = blockIdx.z;               // b*KVH + kvh
    const int tx = threadIdx.x & 31;
    const int ty = threadIdx.x >> 5;         // 0..7
    const int b = bk / KVH, kvh = bk % KVH;
#pragma unroll
    for (int i = ty; i < 32; i += 8)
        tile[i][tx] = Vf[((size_t)(b * SS + s0 + i) * KVH + kvh) * HDIM + d0 + tx];
    __syncthreads();
#pragma unroll
    for (int i = ty; i < 32; i += 8)
        Vt[((size_t)bk * HDIM + d0 + i) * SS + s0 + tx] = f2bf(tile[tx][i]);
}

// ---------------- bf16 WMMA GEMM: C[M,N] f32 = A[M,K] @ B[K,N] ----------------
// Block = 256 threads (8 waves); block tile 256(M) x 64(N); wave = 32x64 strip
// (2 A-frags x 4 B-frags = 8 WMMAs per 32-deep K chunk).
// B tile staged in LDS transposed (BT[n][k], stride 48): fragment reads are
// 2x ds_load_b128 per lane. All fragments loaded before the WMMA burst.
#define BT_LD 48
__global__ void __launch_bounds__(256)
gemm_bf16_kernel(const unsigned short* __restrict__ A,
                 const unsigned short* __restrict__ Bm,
                 float* __restrict__ C, int M, int N, int Kd) {
    __shared__ unsigned short BT[64][BT_LD];
    const int tid  = threadIdx.x;
    const int lane = tid & 31;
    const int wave = tid >> 5;
    const int m0 = (blockIdx.x * 8 + wave) * 32;
    const int n0 = blockIdx.y * 64;
    const unsigned short* Ap0 = A + (size_t)m0 * Kd;
    const unsigned short* Ap1 = Ap0 + (size_t)16 * Kd;

    // coop-fill mapping: 32 rows(k) x 8 threads/row, 8 cols each (one uint4)
    const int fk = tid >> 3;             // 0..31
    const int fc = (tid & 7) << 3;       // 0,8,...,56

    v8f acc[2][4];
#pragma unroll
    for (int g = 0; g < 2; ++g)
#pragma unroll
        for (int t = 0; t < 4; ++t) acc[g][t] = zero8();

    for (int k0 = 0; k0 < Kd; k0 += 32) {
        const unsigned short* src = Bm + (size_t)(k0 + fk) * N + n0 + fc;
        uint4 r = *(const uint4*)src;
        if (k0 + 32 < Kd) {
            __builtin_prefetch(src + (size_t)32 * N, 0, 1);
            __builtin_prefetch(Ap0 + (size_t)(lane & 15) * Kd + k0 + 32, 0, 1);
            __builtin_prefetch(Ap1 + (size_t)(lane & 15) * Kd + k0 + 32, 0, 1);
        }
        __syncthreads();                 // previous chunk's LDS reads done
        union { uint4 q; unsigned short u[8]; } t;
        t.q = r;
#pragma unroll
        for (int j = 0; j < 8; ++j) BT[fc + j][fk] = t.u[j];
        __syncthreads();                 // BT tile visible

        // load ALL fragments first so loads clause together and the
        // 8 WMMAs issue back-to-back behind a single wait.
        bfx16 af0 = load_frag_a(Ap0, Kd, k0);
        bfx16 af1 = load_frag_a(Ap1, Kd, k0);
        bfx16 b0 = load_frag_b_kcontig(&BT[0][0], BT_LD, 0,  0);
        bfx16 b1 = load_frag_b_kcontig(&BT[0][0], BT_LD, 16, 0);
        bfx16 b2 = load_frag_b_kcontig(&BT[0][0], BT_LD, 32, 0);
        bfx16 b3 = load_frag_b_kcontig(&BT[0][0], BT_LD, 48, 0);
        acc[0][0] = wmma_bf16(af0, b0, acc[0][0]);
        acc[0][1] = wmma_bf16(af0, b1, acc[0][1]);
        acc[0][2] = wmma_bf16(af0, b2, acc[0][2]);
        acc[0][3] = wmma_bf16(af0, b3, acc[0][3]);
        acc[1][0] = wmma_bf16(af1, b0, acc[1][0]);
        acc[1][1] = wmma_bf16(af1, b1, acc[1][1]);
        acc[1][2] = wmma_bf16(af1, b2, acc[1][2]);
        acc[1][3] = wmma_bf16(af1, b3, acc[1][3]);
    }
    const int c  = lane & 15;
    const int rb = m0 + ((lane >> 4) << 3);
#pragma unroll
    for (int g = 0; g < 2; ++g) {
#pragma unroll
        for (int i = 0; i < 8; ++i) {
            float* crow = C + (size_t)(rb + g * 16 + i) * N + n0;
            crow[c]      = acc[g][0][i];
            crow[16 + c] = acc[g][1][i];
            crow[32 + c] = acc[g][2][i];
            crow[48 + c] = acc[g][3][i];
        }
    }
}

// ---------------- flash attention (causal, GQA) ----------------
// grid (S/64, H, B), block 128 (4 waves); wave owns 16 query rows.
__global__ void __launch_bounds__(128)
fa_kernel(const unsigned short* __restrict__ Q, const unsigned short* __restrict__ K,
          const unsigned short* __restrict__ Vt, unsigned short* __restrict__ O) {
    __shared__ unsigned short Plds[4][16][32];   // per-wave P tile (bf16)

    const int lane = threadIdx.x & 31;
    const int wave = threadIdx.x >> 5;
    const int b = blockIdx.z;
    const int h = blockIdx.y;
    const int q0 = blockIdx.x * 64 + wave * 16;
    const int kvh = h >> 2;                       // N_REP = H/KV = 4
    const int ldq = HH * HDIM;                    // 2048
    const int ldk = KVH * HDIM;                   // 512

    const unsigned short* Qp  = Q  + ((size_t)(b * SS + q0) * HH + h) * HDIM;
    const unsigned short* Kp  = K  + ((size_t)b * SS * KVH + kvh) * HDIM;
    const unsigned short* Vtp = Vt + ((size_t)(b * KVH + kvh) * HDIM) * SS;

    bfx16 aq0 = load_frag_a(Qp, ldq, 0);
    bfx16 aq1 = load_frag_a(Qp, ldq, 32);

    v8f o0 = zero8(), o1 = zero8(), o2 = zero8(), o3 = zero8();
    float mrow[8], lrow[8];
#pragma unroll
    for (int i = 0; i < 8; ++i) { mrow[i] = -1e30f; lrow[i] = 0.0f; }

    const int half = lane >> 4;
    const int col  = lane & 15;
    const float sc = 0.125f;                      // 1/sqrt(64)

    const int nkt = (q0 + 16 + 31) >> 5;          // causal: 32-key tiles
    for (int kt = 0; kt < nkt; ++kt) {
        const int key0 = kt * 32;
        if (kt + 1 < nkt) {
            __builtin_prefetch(Kp + (size_t)(key0 + 32 + (lane & 15)) * ldk, 0, 1);
            __builtin_prefetch(Vtp + (size_t)(lane & 15) * SS + key0 + 32, 0, 1);
        }

        // K^T fragments first (one clause), then the 4 score WMMAs
        bfx16 bk0 = load_frag_b_kcontig(Kp, ldk, key0,      0);
        bfx16 bk1 = load_frag_b_kcontig(Kp, ldk, key0,      32);
        bfx16 bk2 = load_frag_b_kcontig(Kp, ldk, key0 + 16, 0);
        bfx16 bk3 = load_frag_b_kcontig(Kp, ldk, key0 + 16, 32);
        v8f s0 = zero8(), s1 = zero8();
        s0 = wmma_bf16(aq0, bk0, s0);
        s0 = wmma_bf16(aq1, bk1, s0);
        s1 = wmma_bf16(aq0, bk2, s1);
        s1 = wmma_bf16(aq1, bk3, s1);

        float p0[8], p1[8];
#pragma unroll
        for (int i = 0; i < 8; ++i) {
            const int qrow = q0 + i + (half << 3);
            float x0 = (key0 + col <= qrow)      ? s0[i] * sc : -1e30f;
            float x1 = (key0 + 16 + col <= qrow) ? s1[i] * sc : -1e30f;
            float t = fmaxf(x0, x1);
#pragma unroll
            for (int msk = 1; msk < 16; msk <<= 1)
                t = fmaxf(t, __shfl_xor(t, msk, 32));
            const float mnew = fmaxf(mrow[i], t);
            const float fs = __expf(mrow[i] - mnew);
            const float e0 = __expf(x0 - mnew);
            const float e1 = __expf(x1 - mnew);
            float rs = e0 + e1;
#pragma unroll
            for (int msk = 1; msk < 16; msk <<= 1)
                rs += __shfl_xor(rs, msk, 32);
            lrow[i] = lrow[i] * fs + rs;
            mrow[i] = mnew;
            o0[i] *= fs; o1[i] *= fs; o2[i] *= fs; o3[i] *= fs;
            p0[i] = e0; p1[i] = e1;
        }
        // route P through per-wave LDS tile to re-fragment into A layout
#pragma unroll
        for (int i = 0; i < 8; ++i) {
            Plds[wave][i + (half << 3)][col]      = f2bf(p0[i]);
            Plds[wave][i + (half << 3)][16 + col] = f2bf(p1[i]);
        }
        asm volatile("s_wait_dscnt 0x0" ::: "memory");
        // V fragments + P fragment first, then the 4 PV WMMAs
        bfx16 ap = load_frag_a(&Plds[wave][0][0], 32, 0);
        bfx16 bv0 = load_frag_b_kcontig(Vtp, SS, 0,  key0);
        bfx16 bv1 = load_frag_b_kcontig(Vtp, SS, 16, key0);
        bfx16 bv2 = load_frag_b_kcontig(Vtp, SS, 32, key0);
        bfx16 bv3 = load_frag_b_kcontig(Vtp, SS, 48, key0);
        o0 = wmma_bf16(ap, bv0, o0);
        o1 = wmma_bf16(ap, bv1, o1);
        o2 = wmma_bf16(ap, bv2, o2);
        o3 = wmma_bf16(ap, bv3, o3);
        asm volatile("" ::: "memory");
    }

    unsigned short* Op = O + ((size_t)(b * SS + q0) * HH + h) * HDIM;
#pragma unroll
    for (int i = 0; i < 8; ++i) {
        const float inv = 1.0f / lrow[i];
        const int r = i + (half << 3);
        unsigned short* prow = Op + (size_t)r * ldq;
        prow[col]      = f2bf(o0[i] * inv);
        prow[16 + col] = f2bf(o1[i] * inv);
        prow[32 + col] = f2bf(o2[i] * inv);
        prow[48 + col] = f2bf(o3[i] * inv);
    }
}

// ---------------- launcher ----------------

extern "C" void kernel_launch(void* const* d_in, const int* in_sizes, int n_in,
                              void* d_out, int out_size, void* d_ws, size_t ws_size,
                              hipStream_t stream) {
    const float* x    = (const float*)d_in[0];
    const float* wq   = (const float*)d_in[1];
    const float* wk   = (const float*)d_in[2];
    const float* wv   = (const float*)d_in[3];
    const float* wo   = (const float*)d_in[4];
    const float* fcos = (const float*)d_in[5];
    const float* fsin = (const float*)d_in[6];

    char* w = (char*)d_ws;
    size_t off = 0;
    auto alloc = [&](size_t bytes) -> void* {
        void* p = w + off;
        off = (off + bytes + 255) & ~(size_t)255;
        return p;
    };

    unsigned short* xb   = (unsigned short*)alloc((size_t)MROWS * DD * 2);
    unsigned short* wqb  = (unsigned short*)alloc((size_t)DD * HH * HDIM * 2);
    unsigned short* wkb  = (unsigned short*)alloc((size_t)DD * KVH * HDIM * 2);
    unsigned short* wvb  = (unsigned short*)alloc((size_t)DD * KVH * HDIM * 2);
    unsigned short* wob  = (unsigned short*)alloc((size_t)HH * HDIM * DD * 2);
    float* qf            = (float*)alloc((size_t)MROWS * HH * HDIM * 4);
    float* kf            = (float*)alloc((size_t)MROWS * KVH * HDIM * 4);
    float* vf            = (float*)alloc((size_t)MROWS * KVH * HDIM * 4);
    unsigned short* qbf  = (unsigned short*)alloc((size_t)MROWS * HH * HDIM * 2);
    unsigned short* kbf  = (unsigned short*)alloc((size_t)MROWS * KVH * HDIM * 2);
    unsigned short* vt   = (unsigned short*)alloc((size_t)MROWS * KVH * HDIM * 2);
    unsigned short* attn = (unsigned short*)alloc((size_t)MROWS * HH * HDIM * 2);

    const int TPB = 256;
    auto blocks = [](long long n, int t) { return (unsigned)((n + t - 1) / t); };

    // 1) convert inputs to bf16
    long long nx = (long long)MROWS * DD;
    conv_f32_bf16_kernel<<<blocks(nx, TPB), TPB, 0, stream>>>(x, xb, (int)nx);
    long long nwq = (long long)DD * HH * HDIM;
    conv_f32_bf16_kernel<<<blocks(nwq, TPB), TPB, 0, stream>>>(wq, wqb, (int)nwq);
    long long nwk = (long long)DD * KVH * HDIM;
    conv_f32_bf16_kernel<<<blocks(nwk, TPB), TPB, 0, stream>>>(wk, wkb, (int)nwk);
    conv_f32_bf16_kernel<<<blocks(nwk, TPB), TPB, 0, stream>>>(wv, wvb, (int)nwk);
    long long nwo = (long long)HH * HDIM * DD;
    conv_f32_bf16_kernel<<<blocks(nwo, TPB), TPB, 0, stream>>>(wo, wob, (int)nwo);

    // 2) projections (bf16 WMMA, f32 out), block tile 256x64
    {
        dim3 g(MROWS / 256, (HH * HDIM) / 64);
        gemm_bf16_kernel<<<g, 256, 0, stream>>>(xb, wqb, qf, MROWS, HH * HDIM, DD);
    }
    {
        dim3 g(MROWS / 256, (KVH * HDIM) / 64);
        gemm_bf16_kernel<<<g, 256, 0, stream>>>(xb, wkb, kf, MROWS, KVH * HDIM, DD);
        gemm_bf16_kernel<<<g, 256, 0, stream>>>(xb, wvb, vf, MROWS, KVH * HDIM, DD);
    }

    // 3) RoPE + bf16 for Q,K; fused transpose+convert for V
    long long qpairs = (long long)MROWS * HH * HDIM / 2;
    rope_bf16_kernel<<<blocks(qpairs, TPB), TPB, 0, stream>>>(qf, fcos, fsin, qbf,
                                                              HH, (int)qpairs);
    long long kpairs = (long long)MROWS * KVH * HDIM / 2;
    rope_bf16_kernel<<<blocks(kpairs, TPB), TPB, 0, stream>>>(kf, fcos, fsin, kbf,
                                                              KVH, (int)kpairs);
    {
        dim3 g(SS / 32, HDIM / 32, BB * KVH);
        transpose_v_kernel<<<g, 256, 0, stream>>>(vf, vt);
    }

    // 4) causal GQA flash attention (bf16 WMMA, f32 softmax)
    {
        dim3 g(SS / 64, HH, BB);
        fa_kernel<<<g, 128, 0, stream>>>(qbf, kbf, vt, attn);
    }

    // 5) output projection -> f32 d_out
    {
        dim3 g(MROWS / 256, DD / 64);
        gemm_bf16_kernel<<<g, 256, 0, stream>>>(attn, wob, (float*)d_out,
                                                MROWS, DD, DD);
    }
}
